// CrossAttention_53197464928743
// MI455X (gfx1250) — matrix operand order repaired and checked by reading, not compiled, not run on hardware
//
#include <hip/hip_runtime.h>
#include <hip/hip_bf16.h>

typedef __attribute__((ext_vector_type(16))) _Float16 v16h;
typedef __attribute__((ext_vector_type(8)))  float    v8f;
typedef unsigned int v4u __attribute__((ext_vector_type(4)));
typedef int v8i __attribute__((ext_vector_type(8)));
typedef int v4i __attribute__((ext_vector_type(4)));

#if __has_builtin(__builtin_amdgcn_tensor_load_to_lds)
#define HAVE_TDM 1
#else
#define HAVE_TDM 0
#endif

// ---------------------------------------------------------------------------
// WMMA helpers (gfx1250, wave32).  Layouts used here (A/B in this code's contiguous k order):
//   A (16x32 f16): lane L -> row m = L&15; element e -> K = e + 16*(L>>4)
//   B (32x16 f16): lane L -> col n = L&15; element e -> K = e + 16*(L>>4)
//   C/D (16x16 f32): VGPR r, lane L -> row m = r + 8*(L>>4), col n = L&15
// ---------------------------------------------------------------------------
__device__ inline v8f wmma_f16(v16h a, v16h b, v8f c) {
  return __builtin_amdgcn_wmma_f32_16x16x32_f16(false, a, false, b, (short)0, c,
                                                false, false);
}

__device__ inline v16h load_a_frag(const _Float16* __restrict__ src, int row0,
                                   int ld, int k0, int lane) {
  const int m = lane & 15, half = lane >> 4;
  const _Float16* p = src + (size_t)(row0 + m) * ld + k0;
  v16h a;
#pragma unroll
  for (int e = 0; e < 16; ++e) a[e] = p[(half << 4) + e];
  return a;
}

__device__ inline v16h load_b_frag(const _Float16* __restrict__ src, int col0,
                                   int ld, int k0, int lane) {
  const int n = lane & 15, half = lane >> 4;
  const _Float16* p = src + (size_t)(col0 + n) * ld + k0 + (half << 4);
  v16h b;
#pragma unroll
  for (int e = 0; e < 16; ++e) b[e] = p[e];
  return b;
}

// ---------------------------------------------------------------------------
// Tensor Data Mover: 2-D tile load (global -> LDS), D# per CDNA5 ISA 8.3/8.4.
// data_size = 2 bytes.  One valid descriptor, no multicast / pad / iterate.
// ---------------------------------------------------------------------------
#if HAVE_TDM
__device__ inline void tdm_load_2d(unsigned lds_byte_addr, const void* gsrc,
                                   unsigned tile0, unsigned tile1,
                                   unsigned tensor_d0, unsigned tensor_d1,
                                   unsigned stride0) {
  const unsigned long long ga = (unsigned long long)(size_t)gsrc;
  v4u g0;
  g0.x = 1u;                                    // count=1, user descriptor
  g0.y = lds_byte_addr;                         // D#.lds_addr (bytes)
  g0.z = (unsigned)ga;                          // global_addr[31:0]  (bit 64)
  g0.w = (unsigned)((ga >> 32) & 0x01FFFFFFull) // global_addr[56:32]
         | (2u << 30);                          // type = 2 ("image")
  v8i g1;
  g1[0] = (int)(1u << 16);                      // data_size=1 -> 2 bytes
  g1[1] = (int)((tensor_d0 & 0xFFFFu) << 16);   // tensor_dim0[15:0] @bit48
  g1[2] = (int)((tensor_d0 >> 16) | ((tensor_d1 & 0xFFFFu) << 16));
  g1[3] = (int)((tensor_d1 >> 16) | ((tile0 & 0xFFFFu) << 16)); // tile_dim0
  g1[4] = (int)(tile1 & 0xFFFFu);               // tile_dim1, tile_dim2=0
  g1[5] = (int)stride0;                         // tensor_dim0_stride[31:0]
  g1[6] = 0;
  g1[7] = 0;
  const v4i gz = {0, 0, 0, 0};
#if defined(__clang_major__) && (__clang_major__ >= 23)
  const v8i z8 = {0, 0, 0, 0, 0, 0, 0, 0};
  __builtin_amdgcn_tensor_load_to_lds(g0, g1, gz, gz, z8, 0);
#else
  __builtin_amdgcn_tensor_load_to_lds(g0, g1, gz, gz, 0);
#endif
}
#endif

// ---------------------------------------------------------------------------
// Prep kernels: gather NCHW -> token-major, (pool), LayerNorm(C=256), emit f16
// ---------------------------------------------------------------------------
__global__ void prep_q_kernel(const float* __restrict__ qf,
                              const float* __restrict__ w,
                              const float* __restrict__ bia,
                              _Float16* __restrict__ out) {
  __shared__ float tile[256 * 33];
  const int bid = blockIdx.x;
  const int b = bid >> 7;
  const int hw0 = (bid & 127) << 5;
  const int tid = threadIdx.x;
  const int lane = tid & 31, wv = tid >> 5;
#pragma unroll
  for (int i = 0; i < 32; ++i) {
    const int c = (wv << 5) + i;
    tile[c * 33 + lane] = qf[((size_t)(b * 256 + c) << 12) + hw0 + lane];
  }
  __syncthreads();
  const int tok = tid >> 3, g = tid & 7;
  float vals[32];
  float s = 0.f;
#pragma unroll
  for (int k = 0; k < 32; ++k) {
    vals[k] = tile[(g + (k << 3)) * 33 + tok];
    s += vals[k];
  }
#pragma unroll
  for (int m = 1; m < 8; m <<= 1) s += __shfl_xor(s, m, 32);
  const float mu = s * (1.0f / 256.0f);
  float v = 0.f;
#pragma unroll
  for (int k = 0; k < 32; ++k) { float d = vals[k] - mu; v += d * d; }
#pragma unroll
  for (int m = 1; m < 8; m <<= 1) v += __shfl_xor(v, m, 32);
  const float rstd = rsqrtf(v * (1.0f / 256.0f) + 1e-5f);
  const size_t ob = ((size_t)(b * 4096 + hw0 + tok)) << 8;
#pragma unroll
  for (int k = 0; k < 32; ++k) {
    const int c = g + (k << 3);
    out[ob + c] = (_Float16)((vals[k] - mu) * rstd * w[c] + bia[c]);
  }
}

__global__ void prep_kv_kernel(const float* __restrict__ kvf,
                               const float* __restrict__ w,
                               const float* __restrict__ bia,
                               _Float16* __restrict__ out) {
  __shared__ float tile[256 * 33];
  const int bid = blockIdx.x;
  const int b = bid >> 5;
  const int hk = bid & 31;
  const int tid = threadIdx.x;
  const int lane = tid & 31, wv = tid >> 5;
#pragma unroll
  for (int i = 0; i < 32; ++i) {
    const int c = (wv << 5) + i;
    const float* base = kvf + ((size_t)(b * 256 + c) << 12) + (hk << 7);
    const float2 p0 = *(const float2*)(base + (lane << 1));
    const float2 p1 = *(const float2*)(base + 64 + (lane << 1));
    tile[c * 33 + lane] = 0.25f * (p0.x + p0.y + p1.x + p1.y);
  }
  __syncthreads();
  const int tok = tid >> 3, g = tid & 7;
  float vals[32];
  float s = 0.f;
#pragma unroll
  for (int k = 0; k < 32; ++k) {
    vals[k] = tile[(g + (k << 3)) * 33 + tok];
    s += vals[k];
  }
#pragma unroll
  for (int m = 1; m < 8; m <<= 1) s += __shfl_xor(s, m, 32);
  const float mu = s * (1.0f / 256.0f);
  float v = 0.f;
#pragma unroll
  for (int k = 0; k < 32; ++k) { float d = vals[k] - mu; v += d * d; }
#pragma unroll
  for (int m = 1; m < 8; m <<= 1) v += __shfl_xor(v, m, 32);
  const float rstd = rsqrtf(v * (1.0f / 256.0f) + 1e-5f);
  const size_t ob = ((size_t)(b * 1024 + hk * 32 + tok)) << 8;
#pragma unroll
  for (int k = 0; k < 32; ++k) {
    const int c = g + (k << 3);
    out[ob + c] = (_Float16)((vals[k] - mu) * rstd * w[c] + bia[c]);
  }
}

__global__ void cvt_kernel(const float* __restrict__ src,
                           _Float16* __restrict__ dst, int n) {
  const int i = blockIdx.x * blockDim.x + threadIdx.x;
  if (i < n) dst[i] = (_Float16)src[i];
}

// ---------------------------------------------------------------------------
// Projection GEMM: out = A(M x 256 f16) @ W^T + bias.  Wave: 16 rows x 64 cols.
//   mode 0: qh [B,8,4096,32] (x scale)   mode 1: kh [B,8,1024,32]
//   mode 2: vT [B,8,32,1024]             mode 3: fp32 NCHW output [B,256,64,64]
// ---------------------------------------------------------------------------
__global__ void gemm_kernel(const _Float16* __restrict__ A,
                            const _Float16* __restrict__ W,
                            const float* __restrict__ bias,
                            void* __restrict__ outp,
                            int Mtot, int mode, float scale) {
  const int lane = threadIdx.x & 31;
  const int half = lane >> 4;
  const int wave = blockIdx.x * (blockDim.x >> 5) + (threadIdx.x >> 5);
  const int rt = wave >> 2;
  const int ct = wave & 3;
  if ((rt << 4) >= Mtot) return;
  const int row0 = rt << 4;
  const int col0 = ct << 6;
  v8f acc[4];
#pragma unroll
  for (int t = 0; t < 4; ++t) acc[t] = v8f{0, 0, 0, 0, 0, 0, 0, 0};
#pragma unroll
  for (int kc = 0; kc < 8; ++kc) {
    const v16h a = load_a_frag(A, row0, 256, kc << 5, lane);
#pragma unroll
    for (int t = 0; t < 4; ++t) {
      const v16h bf = load_b_frag(W, col0 + (t << 4), 256, kc << 5, lane);
      acc[t] = wmma_f16(a, bf, acc[t]);
    }
  }
#pragma unroll
  for (int t = 0; t < 4; ++t) {
#pragma unroll
    for (int r = 0; r < 8; ++r) {
      const int tg = row0 + r + (half << 3);
      const int o = col0 + (t << 4) + (lane & 15);
      const float val = (acc[t][r] + bias[o]) * scale;
      if (mode == 0) {
        const int b = tg >> 12, n = tg & 4095, h = o >> 5, d = o & 31;
        ((_Float16*)outp)[((size_t)((b * 8 + h) * 4096 + n) << 5) + d] = (_Float16)val;
      } else if (mode == 1) {
        const int b = tg >> 10, m = tg & 1023, h = o >> 5, d = o & 31;
        ((_Float16*)outp)[((size_t)((b * 8 + h) * 1024 + m) << 5) + d] = (_Float16)val;
      } else if (mode == 2) {
        const int b = tg >> 10, m = tg & 1023, h = o >> 5, d = o & 31;
        ((_Float16*)outp)[((size_t)((b * 8 + h) * 32 + d) << 10) + m] = (_Float16)val;
      } else {
        const int b = tg >> 12, n = tg & 4095;
        ((float*)outp)[((size_t)(b * 256 + o) << 12) + n] = val;
      }
    }
  }
}

// ---------------------------------------------------------------------------
// Flash attention.  Block = 4 waves sharing one (b,h); K/V staged into LDS in
// 256-key chunks via the Tensor Data Mover (wave 0 issues tensor_load_to_lds,
// waits TENSORcnt, block barriers).  Each wave owns a 16-row q tile and runs
// online softmax; P is bounced through 1KB/wave LDS (in-order DS) to convert
// C-layout -> A-layout.  head_dim 32 == one WMMA K step.
// ---------------------------------------------------------------------------
__global__ void attn_kernel(const _Float16* __restrict__ qh,
                            const _Float16* __restrict__ kh,
                            const _Float16* __restrict__ vT,
                            _Float16* __restrict__ ao) {
  __shared__ __align__(128) _Float16 shK[256 * 32];  // 16 KB: 256 keys x 32d
  __shared__ __align__(128) _Float16 shV[32 * 256];  // 16 KB: 32d x 256 keys
  __shared__ _Float16 pls[4 * 512];                  // P bounce, 1KB/wave
  const int lane = threadIdx.x & 31, half = lane >> 4, j = lane & 15;
  const int wslot = threadIdx.x >> 5;
  const int wid = blockIdx.x * 4 + wslot;
  const int qt = wid & 255;
  const int bh = wid >> 8;                 // b*8 + h (same for all 4 waves)
  const int q0 = qt << 4;
  const _Float16* qp = qh + ((size_t)(bh * 4096 + q0) << 5);
  const _Float16* kp = kh + ((size_t)bh << 15);
  const _Float16* vp = vT + ((size_t)bh << 15);
  _Float16* base_p = &pls[wslot * 512];

  v16h aq;
  {
    const _Float16* p = qp + ((size_t)(lane & 15) << 5);
#pragma unroll
    for (int e = 0; e < 16; ++e) aq[e] = p[(half << 4) + e];
  }
  float mrun[8], lrun[8];
  v8f o0 = {0, 0, 0, 0, 0, 0, 0, 0};
  v8f o1 = {0, 0, 0, 0, 0, 0, 0, 0};
#pragma unroll
  for (int r = 0; r < 8; ++r) { mrun[r] = -__builtin_inff(); lrun[r] = 0.f; }

  for (int mc0 = 0; mc0 < 1024; mc0 += 256) {
#if HAVE_TDM
    if (wslot == 0) {
      // K chunk: 256 rows x 32d, contiguous -> 1-D tile of 8192 f16
      tdm_load_2d((unsigned)(size_t)(void*)shK, kp + ((size_t)mc0 << 5),
                  8192u, 1u, 8192u, 1u, 8192u);
      // V chunk: 32 rows (d) x 256 keys, row stride 1024 f16
      tdm_load_2d((unsigned)(size_t)(void*)shV, vp + mc0,
                  256u, 32u, 1024u, 32u, 1024u);
      __builtin_amdgcn_s_wait_tensorcnt(0);
    }
#else
    {  // cooperative fallback copy (128 threads, uint4)
      const uint4* gK = (const uint4*)(kp + ((size_t)mc0 << 5));
      uint4* sK = (uint4*)shK;
      for (int i = threadIdx.x; i < 1024; i += 128) sK[i] = gK[i];
      uint4* sV = (uint4*)shV;
      for (int i = threadIdx.x; i < 1024; i += 128) {
        const int r = i >> 5, cc = i & 31;
        sV[r * 32 + cc] = *(const uint4*)(vp + (size_t)r * 1024 + mc0 + cc * 8);
      }
    }
#endif
    __syncthreads();

    for (int mcl = 0; mcl < 256; mcl += 32) {
      v16h bk0, bk1;
      {
        const _Float16* p0 = &shK[(mcl + j) * 32 + (half << 4)];
        const _Float16* p1 = &shK[(mcl + 16 + j) * 32 + (half << 4)];
#pragma unroll
        for (int e = 0; e < 16; ++e) { bk0[e] = p0[e]; bk1[e] = p1[e]; }
      }
      const v8f z = {0, 0, 0, 0, 0, 0, 0, 0};
      v8f s0 = wmma_f16(aq, bk0, z);
      v8f s1 = wmma_f16(aq, bk1, z);
#pragma unroll
      for (int r = 0; r < 8; ++r) {
        const float x0 = s0[r], x1 = s1[r];
        float cm = fmaxf(x0, x1);
#pragma unroll
        for (int m = 1; m < 16; m <<= 1) cm = fmaxf(cm, __shfl_xor(cm, m, 32));
        const float mn = fmaxf(mrun[r], cm);
        const float alpha = __expf(mrun[r] - mn);
        const float p0 = __expf(x0 - mn);
        const float p1 = __expf(x1 - mn);
        float rs = p0 + p1;
#pragma unroll
        for (int m = 1; m < 16; m <<= 1) rs += __shfl_xor(rs, m, 32);
        lrun[r] = lrun[r] * alpha + rs;
        mrun[r] = mn;
        o0[r] *= alpha;
        o1[r] *= alpha;
        const int row = r + (half << 3);
        base_p[row * 32 + j] = (_Float16)p0;
        base_p[row * 32 + 16 + j] = (_Float16)p1;
      }
      v16h ap;
      {
        const _Float16* p = base_p + ((lane & 15) << 5);
#pragma unroll
        for (int e = 0; e < 16; ++e) ap[e] = p[(half << 4) + e];
      }
      v16h bv0, bv1;
      {
        const _Float16* p0 = &shV[j * 256 + mcl + (half << 4)];
        const _Float16* p1 = &shV[(16 + j) * 256 + mcl + (half << 4)];
#pragma unroll
        for (int e = 0; e < 16; ++e) { bv0[e] = p0[e]; bv1[e] = p1[e]; }
      }
      o0 = wmma_f16(ap, bv0, o0);
      o1 = wmma_f16(ap, bv1, o1);
    }
    __syncthreads();  // before next chunk overwrites shK/shV
  }
  const int b = bh >> 3, h = bh & 7;
#pragma unroll
  for (int r = 0; r < 8; ++r) {
    const float inv = 1.0f / lrun[r];
    const int tok = q0 + r + (half << 3);
    const size_t rowb = ((size_t)(b * 4096 + tok) << 8) + (h << 5);
    ao[rowb + j] = (_Float16)(o0[r] * inv);
    ao[rowb + 16 + j] = (_Float16)(o1[r] * inv);
  }
}

// ---------------------------------------------------------------------------
extern "C" void kernel_launch(void* const* d_in, const int* in_sizes, int n_in,
                              void* d_out, int out_size, void* d_ws, size_t ws_size,
                              hipStream_t stream) {
  (void)in_sizes; (void)n_in; (void)out_size; (void)ws_size;
  const float* q_feat = (const float*)d_in[0];
  const float* kv_feat = (const float*)d_in[1];
  const float* nqw = (const float*)d_in[2];
  const float* nqb = (const float*)d_in[3];
  const float* nkw = (const float*)d_in[4];
  const float* nkb = (const float*)d_in[5];
  const float* wq = (const float*)d_in[6];
  const float* bq = (const float*)d_in[7];
  const float* wk = (const float*)d_in[8];
  const float* bk = (const float*)d_in[9];
  const float* wv = (const float*)d_in[10];
  const float* bv = (const float*)d_in[11];
  const float* wp = (const float*)d_in[12];
  const float* bp = (const float*)d_in[13];

  char* ws = (char*)d_ws;
  _Float16* qn   = (_Float16*)(ws + 0);          // 8192 x 256
  _Float16* kvn  = (_Float16*)(ws + 4194304);    // 2048 x 256
  _Float16* wq16 = (_Float16*)(ws + 5242880);    // 4 x 256x256
  _Float16* wk16 = wq16 + 65536;
  _Float16* wv16 = wk16 + 65536;
  _Float16* wp16 = wv16 + 65536;
  _Float16* qhb  = (_Float16*)(ws + 5767168);    // [2,8,4096,32]
  _Float16* khb  = (_Float16*)(ws + 9961472);    // [2,8,1024,32]
  _Float16* vTb  = (_Float16*)(ws + 11010048);   // [2,8,32,1024]
  _Float16* aob  = (_Float16*)(ws + 12058624);   // 8192 x 256

  prep_q_kernel<<<256, 256, 0, stream>>>(q_feat, nqw, nqb, qn);
  prep_kv_kernel<<<64, 256, 0, stream>>>(kv_feat, nkw, nkb, kvn);
  cvt_kernel<<<256, 256, 0, stream>>>(wq, wq16, 65536);
  cvt_kernel<<<256, 256, 0, stream>>>(wk, wk16, 65536);
  cvt_kernel<<<256, 256, 0, stream>>>(wv, wv16, 65536);
  cvt_kernel<<<256, 256, 0, stream>>>(wp, wp16, 65536);

  const float SCALE = 0.17677669529663687f;  // 32^-0.5, folded into Q proj
  gemm_kernel<<<512, 128, 0, stream>>>(qn,  wq16, bq, (void*)qhb, 8192, 0, SCALE);
  gemm_kernel<<<128, 128, 0, stream>>>(kvn, wk16, bk, (void*)khb, 2048, 1, 1.0f);
  gemm_kernel<<<128, 128, 0, stream>>>(kvn, wv16, bv, (void*)vTb, 2048, 2, 1.0f);
  attn_kernel<<<1024, 128, 0, stream>>>(qhb, khb, vTb, aob);
  gemm_kernel<<<512, 128, 0, stream>>>(aob, wp16, bp, d_out, 8192, 3, 1.0f);
}
